// PhysicsInformedLNN_11905649344789
// MI455X (gfx1250) — matrix operand order repaired and checked by reading, not compiled, not run on hardware
//
#include <hip/hip_runtime.h>
#include <hip/hip_bf16.h>

typedef __attribute__((ext_vector_type(16))) _Float16 v16h;
typedef __attribute__((ext_vector_type(8)))  float    v8f;

#define B_TOTAL 4096
#define T_STEPS 1024
#define HDIM    64
#define DT_C    0.1f
#define ALPHA_C 0.1f

__device__ __forceinline__ float fast_rcp(float x) {
    return __builtin_amdgcn_rcpf(x);      // single v_rcp_f32, no IEEE fixup
}

__device__ __forceinline__ float fast_tanh(float x) {
#if __has_builtin(__builtin_amdgcn_tanhf)
    return __builtin_amdgcn_tanhf(x);     // gfx1250 v_tanh_f32 (TRANS32)
#else
    float e = __expf(2.0f * x);
    return 1.0f - 2.0f * fast_rcp(e + 1.0f);
#endif
}

// A-fragment K mapping for 16-bit 16x32 A (contiguous order, matching B):
// lane half h, element e: K = 16h + e
__device__ __forceinline__ int a_kmap(int e, int h) {
    return 16 * h + e;
}

__global__ __launch_bounds__(128) void lnn_scan_kernel(
    const float* __restrict__ I_seq,     // [B, T, 3]
    const float* __restrict__ sigma_seq, // [B, T]
    const float* __restrict__ W0,        // [64, 64]
    const float* __restrict__ Wi,        // [64, 3]
    const float* __restrict__ W1,        // [16, 64]
    const float* __restrict__ b1,        // [16]
    const float* __restrict__ W2,        // [16] (shape [1,16])
    const float* __restrict__ b2,        // [1]
    float* __restrict__ out)             // [2*B]: crack_risk | sigma_residual
{
    // Ping-pong f16 copy of S (16 rows x 64 cols, stride 72 halves = 144B rows,
    // keeps every b128 access 16B aligned and spreads banks).
    __shared__ __align__(16) _Float16 Sbuf[2][16][72];
    __shared__ float Sf32[16][65];
    __shared__ float Hbuf[16][17];

    const int tid  = threadIdx.x;
    const int wave = tid >> 5;       // n-tile (H columns wave*16 .. wave*16+15)
    const int lane = tid & 31;
    const int h    = lane >> 4;      // lane half
    const int ml   = lane & 15;      // batch row within tile (B/C/D lane index)
    const int b0   = blockIdx.x * 16;
    const int m_global = b0 + ml;

    // zero the initial S buffer
    for (int i = tid; i < 16 * 72; i += 128)
        ((_Float16*)Sbuf[0])[i] = (_Float16)0.0f;

    // ---- constant A-fragments: W0 (two K-blocks) and zero-padded Wi ----
    // Computing Z^T = W0 * S^T, so A row index = output H column.
    v16h w0A0, w0A1, wiA;
    #pragma unroll
    for (int e = 0; e < 16; ++e) {
        const int k   = a_kmap(e, h);
        const int row = wave * 16 + ml;      // A lane holds M = lane%16
        w0A0[e] = (_Float16)W0[row * 64 + k];
        w0A1[e] = (_Float16)W0[row * 64 + 32 + k];
        wiA[e]  = (k < 3) ? (_Float16)Wi[row * 3 + k] : (_Float16)0.0f;
    }

    // persistent fp32 state: this wave's D-tile, element r -> S[m=ml][n=wave*16+r+8h]
    float Sreg[8];
    #pragma unroll
    for (int r = 0; r < 8; ++r) Sreg[r] = 0.0f;

    const float* Irow = I_seq     + (size_t)m_global * T_STEPS * 3;
    const float* srow = sigma_seq + (size_t)m_global * T_STEPS;

    __syncthreads();

    for (int t = 0; t < T_STEPS; ++t) {
        const int cur = t & 1;
        const int nxt = cur ^ 1;

        if (t + 16 < T_STEPS)
            __builtin_prefetch(&Irow[(size_t)(t + 16) * 3], 0, 0);

        // per-batch-row inputs (lane-scalar because B/C lanes index the row)
        const float i0 = Irow[t * 3 + 0];
        const float i1 = Irow[t * 3 + 1];
        const float i2 = Irow[t * 3 + 2];
        const float a  = __expf(-ALPHA_C * srow[t]);

        // B-fragment of I^T (32x16, only K=0..2 nonzero): element e holds K=16h+e
        v16h iB;
        #pragma unroll
        for (int e = 0; e < 16; ++e) iB[e] = (_Float16)0.0f;
        if (h == 0) {
            iB[0] = (_Float16)i0; iB[1] = (_Float16)i1; iB[2] = (_Float16)i2;
        }

        // B-fragments of S^T: lane holds S[m=ml][K], K = kb*32 + 16h + e -> contiguous
        const _Float16* srcRow = &Sbuf[cur][ml][0];
        union V16 { uint4 q[2]; v16h v; };
        V16 sB0, sB1;
        sB0.q[0] = *(const uint4*)(srcRow + 16 * h);
        sB0.q[1] = *(const uint4*)(srcRow + 16 * h + 8);
        sB1.q[0] = *(const uint4*)(srcRow + 32 + 16 * h);
        sB1.q[1] = *(const uint4*)(srcRow + 32 + 16 * h + 8);

        // Z^T tile = W0 * S^T  (K=64 via two chained WMMAs)
        v8f acc = {};
        acc = __builtin_amdgcn_wmma_f32_16x16x32_f16(false, w0A0, false, sB0.v,
                                                     (short)0, acc, false, false);
        acc = __builtin_amdgcn_wmma_f32_16x16x32_f16(false, w0A1, false, sB1.v,
                                                     (short)0, acc, false, false);
        // ip^T tile = Wi * I^T
        v8f ipacc = {};
        ipacc = __builtin_amdgcn_wmma_f32_16x16x32_f16(false, wiA, false, iB,
                                                       (short)0, ipacc, false, false);

        // S <- S + DT*(tanh(a*Z + ip) - S); write f16 copy for next step
        union PK { uint4 q; _Float16 hh[8]; } pk;
        #pragma unroll
        for (int r = 0; r < 8; ++r) {
            const float arg = fmaf(a, acc[r], ipacc[r]);
            const float sn  = Sreg[r] + DT_C * (fast_tanh(arg) - Sreg[r]);
            Sreg[r] = sn;
            pk.hh[r] = (_Float16)sn;
        }
        // D element r is (m=ml, n=wave*16 + 8h + r): 8 contiguous halves -> one b128
        *(uint4*)(&Sbuf[nxt][ml][wave * 16 + 8 * h]) = pk.q;

        __syncthreads();
    }

    // ---- head: h = relu(S @ W1^T + b1); out = h @ W2^T + b2 ----
    #pragma unroll
    for (int r = 0; r < 8; ++r)
        Sf32[ml][wave * 16 + r + 8 * h] = Sreg[r];
    __syncthreads();

    {
        const int m  = tid & 15;
        const int j0 = tid >> 4;          // 0..7, handle j0 and j0+8
        #pragma unroll
        for (int jj = 0; jj < 2; ++jj) {
            const int j = j0 + 8 * jj;
            float s = b1[j];
            #pragma unroll 8
            for (int n = 0; n < HDIM; ++n)
                s = fmaf(Sf32[m][n], W1[j * HDIM + n], s);
            Hbuf[m][j] = fmaxf(s, 0.0f);
        }
    }
    __syncthreads();

    if (tid < 16) {
        const int m = tid;
        float sr = b2[0];
        #pragma unroll
        for (int j = 0; j < 16; ++j)
            sr = fmaf(Hbuf[m][j], W2[j], sr);
        const float crack = fast_rcp(1.0f + __expf(-sr * (1.0f / 500.0f)));
        out[b0 + m]           = crack;
        out[B_TOTAL + b0 + m] = sr;
    }
}

extern "C" void kernel_launch(void* const* d_in, const int* in_sizes, int n_in,
                              void* d_out, int out_size, void* d_ws, size_t ws_size,
                              hipStream_t stream) {
    (void)in_sizes; (void)n_in; (void)d_ws; (void)ws_size; (void)out_size;
    // setup_inputs order:
    // 0 static_params (unused by reference), 1 I_seq, 2 sigma_thermal_seq,
    // 3 W0, 4 Wi, 5 W1, 6 b1, 7 W2, 8 b2
    const float* I_seq = (const float*)d_in[1];
    const float* sigma = (const float*)d_in[2];
    const float* W0    = (const float*)d_in[3];
    const float* Wi    = (const float*)d_in[4];
    const float* W1    = (const float*)d_in[5];
    const float* b1    = (const float*)d_in[6];
    const float* W2    = (const float*)d_in[7];
    const float* b2    = (const float*)d_in[8];
    float* out = (float*)d_out;

    dim3 grid(B_TOTAL / 16);
    dim3 block(128);
    lnn_scan_kernel<<<grid, block, 0, stream>>>(I_seq, sigma, W0, Wi, W1, b1, W2,
                                                b2, out);
}
